// Transformer_41875931136714
// MI455X (gfx1250) — matrix operand order repaired and checked by reading, not compiled, not run on hardware
//
#include <hip/hip_runtime.h>
#include <hip/hip_bf16.h>

// ---------------------------------------------------------------------------
// Model constants (match reference)
// ---------------------------------------------------------------------------
#define BB   2
#define TT   2048
#define CC   1024
#define HH   16
#define DD   64
#define LL   6
#define VV   32000
#define FFN  4096
#define NROW (BB * TT)   // 4096 token rows
#define EPSF 1e-5f

typedef __attribute__((ext_vector_type(16))) __bf16 v16bf;
typedef __attribute__((ext_vector_type(8)))  float  v8f;
typedef unsigned short u16;
typedef unsigned int   u32;

// fp32 -> bf16, round-to-nearest-even
__device__ __forceinline__ u16 f2bf(float f) {
    u32 u = __float_as_uint(f);
    u32 r = (u + 0x7FFFu + ((u >> 16) & 1u)) >> 16;
    return (u16)r;
}

// ---------------------------------------------------------------------------
// CDNA5 async global->LDS copy (ASYNCcnt-tracked; ISA 15.18.3 op 98).
// VDST VGPR holds the LDS byte offset; generic LDS pointers carry the raw
// offset in their low 32 bits (aperture scheme), so the cast is exact.
// ---------------------------------------------------------------------------
__device__ __forceinline__ void async_b128(void* lds_ptr, const void* gptr) {
    const u32 loff = (u32)(size_t)lds_ptr;
    asm volatile("global_load_async_to_lds_b128 %0, %1, off"
                 :: "v"(loff), "v"(gptr) : "memory");
}
__device__ __forceinline__ void wait_async0() {
    asm volatile("s_wait_asynccnt 0" ::: "memory");
}

// xor-shuffle within 16-lane halves via ds_swizzle_b32 (group-of-32 encoding:
// offset[14:10]=xor, offset[4:0]=and). No LDS memory, no address VGPR.
#define SWZ_XOR(x, m) \
    __int_as_float(__builtin_amdgcn_ds_swizzle(__float_as_int(x), (((m) << 10) | 0x1F)))

// ---------------------------------------------------------------------------
// WMMA fragment helpers (contiguous k order k = 16*(lane>>4) + s, same in A and B, wave32)
// ---------------------------------------------------------------------------
// A-matrix 16x32 bf16: lane m=lane&15, kh=lane>>4; VGPR v holds
// K = kh*16 + 2v and +1.
__device__ __forceinline__ v16bf load_a_frag(const u16* base, int ld) {
    union { v16bf v; u16 u[16]; } f;
    const int lane = threadIdx.x & 31;
    const int m    = lane & 15;
    const int kh   = (lane >> 4) & 1;
#pragma unroll
    for (int r = 0; r < 8; ++r) {
        const int k = kh * 16 + 2 * r;
        f.u[2 * r]     = base[m * ld + k];
        f.u[2 * r + 1] = base[m * ld + k + 1];
    }
    return f.v;
}

// B-matrix 32x16 bf16 from row-major [K][N] LDS tile: lane n=lane&15,
// kg=lane>>4; VGPR v holds K = kg*16 + 2v and 2v+1.
__device__ __forceinline__ v16bf load_b_frag(const u16* base, int ld) {
    union { v16bf v; u16 u[16]; } f;
    const int lane = threadIdx.x & 31;
    const int n    = lane & 15;
    const int kg   = (lane >> 4) & 1;
#pragma unroll
    for (int r = 0; r < 8; ++r) {
        const int kk = kg * 16 + 2 * r;
        f.u[2 * r]     = base[kk * ld + n];
        f.u[2 * r + 1] = base[(kk + 1) * ld + n];
    }
    return f.v;
}

// B-matrix 32x16 where memory holds the TRANSPOSE ([N][K], e.g. K-cache rows):
// B[kk][n] = mem[n][kk].
__device__ __forceinline__ v16bf load_bT_frag(const u16* base, int ld) {
    union { v16bf v; u16 u[16]; } f;
    const int lane = threadIdx.x & 31;
    const int n    = lane & 15;
    const int kg   = (lane >> 4) & 1;
#pragma unroll
    for (int r = 0; r < 8; ++r) {
        const int kk = kg * 16 + 2 * r;
        f.u[2 * r]     = base[n * ld + kk];
        f.u[2 * r + 1] = base[n * ld + kk + 1];
    }
    return f.v;
}

__device__ __forceinline__ v8f wmma_bf16(v16bf a, v16bf b, v8f c) {
    return __builtin_amdgcn_wmma_f32_16x16x32_bf16(
        /*neg_a=*/false, a, /*neg_b=*/false, b,
        /*c_mod=*/(short)0, c, /*reuse_a=*/false, /*reuse_b=*/false);
}

// ---------------------------------------------------------------------------
// Embedding: x[b,t,:] = src_emb[src[b,t],:] + pos_emb[t,:]
// ---------------------------------------------------------------------------
__global__ __launch_bounds__(256) void embed_kernel(
    const int* __restrict__ src, const float* __restrict__ src_emb,
    const float* __restrict__ pos_emb, float* __restrict__ x) {
    const size_t i = (size_t)blockIdx.x * 256 + threadIdx.x;  // over NROW*CC
    const int c  = (int)(i & (CC - 1));
    const int bt = (int)(i >> 10);           // CC = 1024 = 2^10
    const int t  = bt & (TT - 1);            // TT = 2048
    const int tok = src[bt];
    x[i] = src_emb[(size_t)tok * CC + c] + pos_emb[(size_t)t * CC + c];
}

// ---------------------------------------------------------------------------
// LayerNorm: fp32 [N,CC] row -> bf16 [N,CC] row. One block per row.
// ---------------------------------------------------------------------------
__global__ __launch_bounds__(256) void layernorm_kernel(
    const float* __restrict__ x, const float* __restrict__ w,
    const float* __restrict__ b, u16* __restrict__ out) {
    __shared__ float red[256];
    const int row = blockIdx.x;
    const int tid = threadIdx.x;
    const float4 vv = ((const float4*)(x + (size_t)row * CC))[tid];

    float s = vv.x + vv.y + vv.z + vv.w;
    red[tid] = s; __syncthreads();
    for (int o = 128; o > 0; o >>= 1) { if (tid < o) red[tid] += red[tid + o]; __syncthreads(); }
    const float mu = red[0] * (1.0f / CC);
    __syncthreads();

    float d0 = vv.x - mu, d1 = vv.y - mu, d2 = vv.z - mu, d3 = vv.w - mu;
    red[tid] = d0 * d0 + d1 * d1 + d2 * d2 + d3 * d3; __syncthreads();
    for (int o = 128; o > 0; o >>= 1) { if (tid < o) red[tid] += red[tid + o]; __syncthreads(); }
    const float rinv = rsqrtf(red[0] * (1.0f / CC) + EPSF);

    const int c0 = tid * 4;
    const float y0 = d0 * rinv * w[c0 + 0] + b[c0 + 0];
    const float y1 = d1 * rinv * w[c0 + 1] + b[c0 + 1];
    const float y2 = d2 * rinv * w[c0 + 2] + b[c0 + 2];
    const float y3 = d3 * rinv * w[c0 + 3] + b[c0 + 3];
    const u32 lo = (u32)f2bf(y0) | ((u32)f2bf(y1) << 16);
    const u32 hi = (u32)f2bf(y2) | ((u32)f2bf(y3) << 16);
    *(uint2*)(out + (size_t)row * CC + c0) = make_uint2(lo, hi);
}

// ---------------------------------------------------------------------------
// GEMM: Y[M,N] = act(Abf16[M,K] @ Wf32[K,N] + bias)
// block = 128 threads (4 waves), tile 128x64, K-step 64.
// Each wave owns a 32x64 strip: 2 A-frags x 4 B-frags x 2 k-substeps
// = 16 WMMA per staging round.
// ACT:  0 = none, 1 = exact GeLU
// MODE: 0 = store f32, 1 = store bf16, 2 = f32 residual accumulate (Out += v)
// ---------------------------------------------------------------------------
template <int ACT, int MODE>
__global__ __launch_bounds__(128) void gemm_bf16_wmma(
    const u16* __restrict__ A, const float* __restrict__ W,
    const float* __restrict__ bias, void* __restrict__ Out,
    int M, int Nn, int Kd) {
    __shared__ u16 As[128][72];  // 128x64 bf16 (+pad)
    __shared__ u16 Bs[64][72];   // 64x64 bf16 (+pad)
    const int tid  = threadIdx.x;
    const int wave = tid >> 5;
    const int lane = tid & 31;
    const int gm0  = blockIdx.y * 128;
    const int gn0  = blockIdx.x * 64;

    v8f acc[2][4] = {};

    for (int k0 = 0; k0 < Kd; k0 += 64) {
        // Stage A tile async: 128 rows x 64 bf16 cols (16B chunks, ASYNCcnt)
#pragma unroll
        for (int it = 0; it < 8; ++it) {
            const int cid = it * 128 + tid;          // 0..1023
            const int row = cid >> 3;
            const int c8  = (cid & 7) * 8;
            async_b128(&As[row][c8],
                       A + (size_t)(gm0 + row) * Kd + k0 + c8);
        }
        // Stage W tile: 64 rows x 64 f32 cols -> bf16 (conversion => VALU path)
#pragma unroll
        for (int it = 0; it < 8; ++it) {
            const int cid = it * 128 + tid;          // 0..1023
            const int row = cid >> 4;
            const int c4  = (cid & 15) * 4;
            const float4 w4 =
                *(const float4*)(W + (size_t)(k0 + row) * Nn + gn0 + c4);
            const u32 lo = (u32)f2bf(w4.x) | ((u32)f2bf(w4.y) << 16);
            const u32 hi = (u32)f2bf(w4.z) | ((u32)f2bf(w4.w) << 16);
            *(uint2*)&Bs[row][c4] = make_uint2(lo, hi);
        }
        if (k0 + 64 < Kd) {  // hint next K-tile toward L2 (global_prefetch_b8)
            __builtin_prefetch(W + (size_t)(k0 + 64 + (tid & 63)) * Nn + gn0, 0, 0);
        }
        wait_async0();       // barrier waits DS/LOAD counters, not ASYNCcnt
        __syncthreads();

#pragma unroll
        for (int kk = 0; kk < 64; kk += 32) {
            const v16bf a0 = load_a_frag(&As[wave * 32][kk], 72);
            const v16bf a1 = load_a_frag(&As[wave * 32 + 16][kk], 72);
#pragma unroll
            for (int j = 0; j < 4; ++j) {
                const v16bf b = load_b_frag(&Bs[kk][16 * j], 72);
                acc[0][j] = wmma_bf16(a0, b, acc[0][j]);
                acc[1][j] = wmma_bf16(a1, b, acc[1][j]);
            }
        }
        __syncthreads();
    }

    // Epilogue: C/D layout -> lanes<16 own rows 0..7, lanes>=16 own rows 8..15
    const int n_in   = lane & 15;
    const int m_base = (lane < 16) ? 0 : 8;
#pragma unroll
    for (int j = 0; j < 4; ++j) {
        const int col = gn0 + 16 * j + n_in;
        const float bcol = bias ? bias[col] : 0.0f;
#pragma unroll
        for (int i = 0; i < 2; ++i) {
#pragma unroll
            for (int r = 0; r < 8; ++r) {
                const int row = gm0 + wave * 32 + i * 16 + m_base + r;
                float v = acc[i][j][r] + bcol;
                if (ACT == 1) v = 0.5f * v * (1.0f + erff(v * 0.70710678118654752f));
                const size_t idx = (size_t)row * Nn + col;
                if (MODE == 0)      ((float*)Out)[idx] = v;
                else if (MODE == 1) ((u16*)Out)[idx]   = f2bf(v);
                else                ((float*)Out)[idx] += v;
            }
        }
    }
}

// ---------------------------------------------------------------------------
// Flash attention (no mask): one block per (b,h, 64-query tile).
// 4 waves x 16 query rows; 64-key S-tiles; online softmax; QK^T and P.V via
// WMMA. q,k,v,o are bf16 [B*T, C] with the head packed at column h*D.
// ---------------------------------------------------------------------------
__global__ __launch_bounds__(128) void flash_attn_kernel(
    const u16* __restrict__ Q, const u16* __restrict__ Kx,
    const u16* __restrict__ Vx, u16* __restrict__ O) {
    __shared__ u16 Qs[64][72];
    __shared__ u16 Ks[64][72];
    __shared__ u16 Vs[64][72];
    __shared__ u16 Ps[4][16][72];   // per-wave P staging (C-layout -> A-layout)

    const int tid  = threadIdx.x;
    const int wave = tid >> 5;
    const int lane = tid & 31;
    const int bh = blockIdx.y;
    const int b  = bh >> 4;               // H = 16
    const int hh = bh & 15;
    const size_t base = (size_t)b * TT * CC + (size_t)hh * DD;
    const int q0 = blockIdx.x * 64;

    // Load the 64x64 Q tile once (async; fenced before first use below)
#pragma unroll
    for (int it = 0; it < 4; ++it) {
        const int cid = it * 128 + tid;   // 0..511
        const int row = cid >> 3;
        const int c8  = (cid & 7) * 8;
        async_b128(&Qs[row][c8], Q + base + (size_t)(q0 + row) * CC + c8);
    }

    const int n_in   = lane & 15;
    const int m_base = (lane < 16) ? 0 : 8;
    float rowm[8], rsum[8];
    v8f accO[4] = {};
#pragma unroll
    for (int r = 0; r < 8; ++r) { rowm[r] = -3.0e38f; rsum[r] = 0.0f; }

    for (int s0 = 0; s0 < TT; s0 += 64) {
        __syncthreads();  // previous-iteration Ks/Vs readers done
#pragma unroll
        for (int it = 0; it < 4; ++it) {
            const int cid = it * 128 + tid;
            const int row = cid >> 3;
            const int c8  = (cid & 7) * 8;
            async_b128(&Ks[row][c8], Kx + base + (size_t)(s0 + row) * CC + c8);
            async_b128(&Vs[row][c8], Vx + base + (size_t)(s0 + row) * CC + c8);
        }
        wait_async0();
        __syncthreads();

        // S = (Q . K^T) * 1/sqrt(D)
        v8f accS[4] = {};
#pragma unroll
        for (int kd = 0; kd < DD; kd += 32) {
            const v16bf aq = load_a_frag(&Qs[wave * 16][kd], 72);
#pragma unroll
            for (int j = 0; j < 4; ++j) {
                const v16bf bk = load_bT_frag(&Ks[16 * j][kd], 72);
                accS[j] = wmma_bf16(aq, bk, accS[j]);
            }
        }

        // online softmax update (rows owned per C/D lane layout)
        float mnew[8];
#pragma unroll
        for (int r = 0; r < 8; ++r) {
            float mx = -3.0e38f;
#pragma unroll
            for (int j = 0; j < 4; ++j) {
                accS[j][r] *= 0.125f;               // 1/sqrt(64)
                mx = fmaxf(mx, accS[j][r]);
            }
            mx = fmaxf(mx, SWZ_XOR(mx, 1));
            mx = fmaxf(mx, SWZ_XOR(mx, 2));
            mx = fmaxf(mx, SWZ_XOR(mx, 4));
            mx = fmaxf(mx, SWZ_XOR(mx, 8));
            mnew[r] = fmaxf(rowm[r], mx);
        }
#pragma unroll
        for (int r = 0; r < 8; ++r) {
            const float alpha = __expf(rowm[r] - mnew[r]);
            rsum[r] *= alpha;
#pragma unroll
            for (int j = 0; j < 4; ++j) accO[j][r] *= alpha;
            float psum = 0.0f;
#pragma unroll
            for (int j = 0; j < 4; ++j) {
                const float p = __expf(accS[j][r] - mnew[r]);
                psum += p;
                Ps[wave][m_base + r][16 * j + n_in] = f2bf(p);
            }
            psum += SWZ_XOR(psum, 1);
            psum += SWZ_XOR(psum, 2);
            psum += SWZ_XOR(psum, 4);
            psum += SWZ_XOR(psum, 8);
            rsum[r] += psum;
            rowm[r] = mnew[r];
        }

        // O += P . V  (P re-loaded in A-layout from this wave's LDS staging)
#pragma unroll
        for (int kk = 0; kk < 64; kk += 32) {
            const v16bf ap = load_a_frag(&Ps[wave][0][kk], 72);
#pragma unroll
            for (int j = 0; j < 4; ++j) {
                const v16bf bv = load_b_frag(&Vs[kk][16 * j], 72);
                accO[j] = wmma_bf16(ap, bv, accO[j]);
            }
        }
    }

    // normalize and store bf16 O
#pragma unroll
    for (int r = 0; r < 8; ++r) {
        const float inv = 1.0f / rsum[r];
        const int row = q0 + wave * 16 + m_base + r;
#pragma unroll
        for (int j = 0; j < 4; ++j) {
            O[base + (size_t)row * CC + 16 * j + n_in] = f2bf(accO[j][r] * inv);
        }
    }
}

// ---------------------------------------------------------------------------
// Host-side launch
// ---------------------------------------------------------------------------
extern "C" void kernel_launch(void* const* d_in, const int* in_sizes, int n_in,
                              void* d_out, int out_size, void* d_ws, size_t ws_size,
                              hipStream_t stream) {
    (void)in_sizes; (void)n_in; (void)out_size; (void)ws_size;
    const int*   src     = (const int*)d_in[0];
    const float* src_emb = (const float*)d_in[1];
    const float* pos_emb = (const float*)d_in[2];
    const float* ln1_w   = (const float*)d_in[3];
    const float* ln1_b   = (const float*)d_in[4];
    const float* wq = (const float*)d_in[5];  const float* bq = (const float*)d_in[6];
    const float* wk = (const float*)d_in[7];  const float* bk = (const float*)d_in[8];
    const float* wv = (const float*)d_in[9];  const float* bv = (const float*)d_in[10];
    const float* wo = (const float*)d_in[11]; const float* bo = (const float*)d_in[12];
    const float* ln2_w = (const float*)d_in[13]; const float* ln2_b = (const float*)d_in[14];
    const float* w1 = (const float*)d_in[15]; const float* b1 = (const float*)d_in[16];
    const float* w2 = (const float*)d_in[17]; const float* b2 = (const float*)d_in[18];
    const float* lnf_w = (const float*)d_in[19]; const float* lnf_b = (const float*)d_in[20];
    const float* head_w = (const float*)d_in[21];
    float* out = (float*)d_out;

    // workspace layout (~88 MiB total)
    char* ws = (char*)d_ws;
    const size_t szX = (size_t)NROW * CC * sizeof(float);   // 16 MiB
    const size_t szH = (size_t)NROW * CC * sizeof(u16);     //  8 MiB
    const size_t szF = (size_t)NROW * FFN * sizeof(u16);    // 32 MiB
    float* x  = (float*)ws;                 ws += szX;
    u16* h    = (u16*)ws;                   ws += szH;
    u16* qb   = (u16*)ws;                   ws += szH;
    u16* kb   = (u16*)ws;                   ws += szH;
    u16* vb   = (u16*)ws;                   ws += szH;
    u16* ob   = (u16*)ws;                   ws += szH;
    u16* ffh  = (u16*)ws;                   ws += szF;

    const dim3 gGemmC(CC / 64, NROW / 128);    // N = 1024 outputs
    const dim3 gGemmF(FFN / 64, NROW / 128);   // N = 4096 outputs
    const dim3 gGemmV(VV / 64, NROW / 128);    // N = 32000 outputs
    const dim3 gAttn(TT / 64, BB * HH);

    embed_kernel<<<(NROW * CC) / 256, 256, 0, stream>>>(src, src_emb, pos_emb, x);

    for (int l = 0; l < LL; ++l) {
        const size_t oCC = (size_t)l * CC * CC;
        const size_t oC  = (size_t)l * CC;
        const size_t oCF = (size_t)l * CC * FFN;
        const size_t oF  = (size_t)l * FFN;

        layernorm_kernel<<<NROW, 256, 0, stream>>>(x, ln1_w + oC, ln1_b + oC, h);
        gemm_bf16_wmma<0, 1><<<gGemmC, 128, 0, stream>>>(h, wq + oCC, bq + oC, qb, NROW, CC, CC);
        gemm_bf16_wmma<0, 1><<<gGemmC, 128, 0, stream>>>(h, wk + oCC, bk + oC, kb, NROW, CC, CC);
        gemm_bf16_wmma<0, 1><<<gGemmC, 128, 0, stream>>>(h, wv + oCC, bv + oC, vb, NROW, CC, CC);
        flash_attn_kernel<<<gAttn, 128, 0, stream>>>(qb, kb, vb, ob);
        gemm_bf16_wmma<0, 2><<<gGemmC, 128, 0, stream>>>(ob, wo + oCC, bo + oC, x, NROW, CC, CC);
        layernorm_kernel<<<NROW, 256, 0, stream>>>(x, ln2_w + oC, ln2_b + oC, h);
        gemm_bf16_wmma<1, 1><<<gGemmF, 128, 0, stream>>>(h, w1 + oCF, b1 + oF, ffh, NROW, FFN, CC);
        gemm_bf16_wmma<0, 2><<<gGemmC, 128, 0, stream>>>(ffh, w2 + oCF, b2 + oC, x, NROW, CC, FFN);
    }

    layernorm_kernel<<<NROW, 256, 0, stream>>>(x, lnf_w, lnf_b, h);
    gemm_bf16_wmma<0, 0><<<gGemmV, 128, 0, stream>>>(h, head_w, nullptr, out, NROW, VV, CC);
}
